// S2S_Net_87746181857739
// MI455X (gfx1250) — compile-verified, matrix operand order checked by reading, not run on hardware
//
#include <hip/hip_runtime.h>
#include <hip/hip_bf16.h>

// ---------------- problem constants (match reference) ----------------
#define BB   128
#define TT   80
#define INF_ 4096
#define HH   512
#define CC   32000
#define LL   10
#define H3   1536   // 3*H

typedef __attribute__((ext_vector_type(16))) __bf16 v16bf;
typedef __attribute__((ext_vector_type(8)))  float  v8f;

#define WMMA_BF16(A, Bm, Cacc) \
  __builtin_amdgcn_wmma_f32_16x16x32_bf16(false, (A), false, (Bm), (short)0, (Cacc), false, false)

// ---- CDNA5 async global->LDS path (ASYNCcnt), with portable fallback ----
#if defined(__HIP_DEVICE_COMPILE__) && __has_builtin(__builtin_amdgcn_global_load_async_to_lds_b128)
#define HAVE_ASYNC 1
typedef int v4i_ __attribute__((__vector_size__(4 * sizeof(int))));
typedef __attribute__((address_space(1))) v4i_ as1_v4i;
typedef __attribute__((address_space(3))) v4i_ as3_v4i;
#else
#define HAVE_ASYNC 0
#endif

__device__ __forceinline__ void cp16_stage(const __bf16* g, __bf16* l) {
#if HAVE_ASYNC
  __builtin_amdgcn_global_load_async_to_lds_b128((as1_v4i*)g, (as3_v4i*)l, 0, 0);
#else
  *reinterpret_cast<int4*>(l) = *reinterpret_cast<const int4*>(g);
#endif
}

__device__ __forceinline__ void wait_async_le(int n) {
#if HAVE_ASYNC
#if __has_builtin(__builtin_amdgcn_s_wait_asynccnt)
  if (n == 0) __builtin_amdgcn_s_wait_asynccnt(0);
  else        __builtin_amdgcn_s_wait_asynccnt(6);
#else
  if (n == 0) asm volatile("s_wait_asynccnt 0x0" ::: "memory");
  else        asm volatile("s_wait_asynccnt 0x6" ::: "memory");
#endif
#else
  (void)n;
#endif
}

// ---------------- fp32 -> bf16 conversion ----------------
__global__ __launch_bounds__(256)
void cvt_f32_bf16(const float* __restrict__ s, __bf16* __restrict__ d, size_t n) {
  size_t i = (size_t)blockIdx.x * 256 + threadIdx.x;
  size_t st = (size_t)gridDim.x * 256;
  for (; i < n; i += st) d[i] = (__bf16)s[i];
}

// encode_out = y1f + y1b (bf16 in, f32 out)
__global__ __launch_bounds__(256)
void add_bf16_f32(const __bf16* __restrict__ a, const __bf16* __restrict__ b,
                  float* __restrict__ o, size_t n) {
  size_t i = (size_t)blockIdx.x * 256 + threadIdx.x;
  size_t st = (size_t)gridDim.x * 256;
  for (; i < n; i += st) o[i] = (float)a[i] + (float)b[i];
}

// ---------------- generic WMMA GEMM: C = act(A[M,K] @ B[N,K]^T + bias) ----------------
// 128x64 tile, 256 threads = 8 waves (4x2), K staged 64 at a time into a
// double-buffered LDS (async loads, ASYNCcnt pipelined against WMMA).
// All call sites: M %128==0, N %64==0, K %64==0 -> no bounds checks.
__global__ __launch_bounds__(256)
void wmma_gemm_kernel(const __bf16* __restrict__ A, const __bf16* __restrict__ Bm,
                      const float* __restrict__ bias,
                      float* __restrict__ outF, __bf16* __restrict__ outB,
                      int M, int N, int K, int out_stride, int relu)
{
  __shared__ __bf16 As[2][128][64];
  __shared__ __bf16 Bs[2][64][64];
  const int m0 = blockIdx.y * 128;
  const int n0 = blockIdx.x * 64;
  const int tid  = threadIdx.x;
  const int lane = tid & 31;
  const int w    = tid >> 5;
  const int wm   = w >> 1;           // 0..3 (M)
  const int wn   = w & 1;            // 0..1 (N)
  const int lr   = lane & 15;
  const int kb   = (lane >> 4) * 8;  // K-half selector per CDNA5 16-bit A/B layout

  v8f acc[2][2] = {};

  const int niter = K >> 6;

  auto issue = [&](int s, int k0) {
    // A stage: 128x64 = 1024 16B chunks -> 4 per thread
#pragma unroll
    for (int i = 0; i < 4; ++i) {
      int c = tid + 256 * i;
      int r = c >> 3, cc = (c & 7) * 8;
      cp16_stage(A + (size_t)(m0 + r) * K + k0 + cc, &As[s][r][cc]);
    }
    // B stage: 64x64 = 512 chunks -> 2 per thread
#pragma unroll
    for (int i = 0; i < 2; ++i) {
      int c = tid + 256 * i;
      int r = c >> 3, cc = (c & 7) * 8;
      cp16_stage(Bm + (size_t)(n0 + r) * K + k0 + cc, &Bs[s][r][cc]);
    }
  };

  issue(0, 0);
  for (int it = 0; it < niter; ++it) {
    const int buf = it & 1;
    if (it + 1 < niter) { issue(buf ^ 1, (it + 1) * 64); wait_async_le(6); }
    else                { wait_async_le(0); }
    __syncthreads();   // stage `buf` visible to all waves

#pragma unroll
    for (int kk = 0; kk < 64; kk += 32) {
      v16bf a0, a1, b0, b1;
#pragma unroll
      for (int i = 0; i < 8; ++i) {
        a0[i]     = As[buf][wm * 32 + lr][kk + kb + i];
        a0[8 + i] = As[buf][wm * 32 + lr][kk + kb + 16 + i];
        a1[i]     = As[buf][wm * 32 + 16 + lr][kk + kb + i];
        a1[8 + i] = As[buf][wm * 32 + 16 + lr][kk + kb + 16 + i];
        b0[i]     = Bs[buf][wn * 32 + lr][kk + kb + i];
        b0[8 + i] = Bs[buf][wn * 32 + lr][kk + kb + 16 + i];
        b1[i]     = Bs[buf][wn * 32 + 16 + lr][kk + kb + i];
        b1[8 + i] = Bs[buf][wn * 32 + 16 + lr][kk + kb + 16 + i];
      }
      acc[0][0] = WMMA_BF16(a0, b0, acc[0][0]);
      acc[0][1] = WMMA_BF16(a0, b1, acc[0][1]);
      acc[1][0] = WMMA_BF16(a1, b0, acc[1][0]);
      acc[1][1] = WMMA_BF16(a1, b1, acc[1][1]);
    }
    __syncthreads();   // done reading `buf` before it is refilled
  }

#pragma unroll
  for (int fr = 0; fr < 2; ++fr) {
#pragma unroll
    for (int fc = 0; fc < 2; ++fc) {
      const int c  = n0 + wn * 32 + fc * 16 + lr;
      const float bv = bias ? bias[c] : 0.f;
#pragma unroll
      for (int i = 0; i < 8; ++i) {
        int r = m0 + wm * 32 + fr * 16 + (lane >> 4) * 8 + i;  // C/D lane layout
        float v = acc[fr][fc][i] + bv;
        if (relu) v = fmaxf(v, 0.f);
        size_t o = (size_t)r * out_stride + c;
        if (outF) outF[o] = v;
        if (outB) outB[o] = (__bf16)v;
      }
    }
  }
}

// ---------------- persistent GRU layer ----------------
// One block owns 16 batch rows for all T steps. h lives in LDS (bf16).
// gi = x@wih^T + bih precomputed (row = b*T + t, width 3H).
// Each wave owns 4 feature tiles; r/z/n accumulators share lane mapping -> in-register gates.
template <bool REVERSE>
__global__ __launch_bounds__(256)
void gru_layer_kernel(const float* __restrict__ gi,    // (B*T, 3H)
                      const __bf16* __restrict__ whh,  // (3H, H) bf16
                      const float* __restrict__ bhh,   // (3H)
                      __bf16* __restrict__ y,          // out hidden per step (bf16)
                      int ystride, int yoff,
                      float* __restrict__ hT_f, __bf16* __restrict__ hT_b)
{
  __shared__ __bf16 hs[16][HH];
  const int b0   = blockIdx.x * 16;
  const int tid  = threadIdx.x;
  const int lane = tid & 31;
  const int w    = tid >> 5;
  const int lr   = lane & 15;
  const int kb   = (lane >> 4) * 8;
  const int bl8  = (lane >> 4) * 8;

  for (int i = tid; i < 16 * HH; i += 256) ((__bf16*)hs)[i] = (__bf16)0.f;
  __syncthreads();

  for (int t = 0; t < TT; ++t) {
    const int tt = REVERSE ? (TT - 1 - t) : t;
    float hnew[4][8];
#pragma unroll
    for (int f = 0; f < 4; ++f) {
      const int c0 = (w * 4 + f) * 16;
      v8f ar = {}, az = {}, an = {};
      for (int k0 = 0; k0 < HH; k0 += 32) {
        v16bf a, br, bz, bn;
        const __bf16* wr  = whh + (size_t)(c0 + lr) * HH + k0 + kb;
        const __bf16* wz  = wr + (size_t)HH * HH;
        const __bf16* wn_ = wr + (size_t)2 * HH * HH;
#pragma unroll
        for (int i = 0; i < 8; ++i) {
          a[i]      = hs[lr][k0 + kb + i];
          a[8 + i]  = hs[lr][k0 + kb + 16 + i];
          br[i] = wr[i];  br[8 + i] = wr[16 + i];
          bz[i] = wz[i];  bz[8 + i] = wz[16 + i];
          bn[i] = wn_[i]; bn[8 + i] = wn_[16 + i];
        }
        ar = WMMA_BF16(a, br, ar);
        az = WMMA_BF16(a, bz, az);
        an = WMMA_BF16(a, bn, an);
      }
      const int j = c0 + lr;
#pragma unroll
      for (int i = 0; i < 8; ++i) {
        const int bl = bl8 + i;
        const size_t grow = ((size_t)(b0 + bl) * TT + tt) * H3;
        float ir  = gi[grow + j];
        float iz  = gi[grow + HH + j];
        float in_ = gi[grow + 2 * HH + j];
        float hr  = ar[i] + bhh[j];
        float hz  = az[i] + bhh[HH + j];
        float hn  = an[i] + bhh[2 * HH + j];
        float rg  = 1.f / (1.f + __expf(-(ir + hr)));
        float zg  = 1.f / (1.f + __expf(-(iz + hz)));
        float ng  = tanhf(in_ + rg * hn);
        float hp  = (float)hs[bl][j];
        hnew[f][i] = (1.f - zg) * ng + zg * hp;
      }
    }
    __syncthreads();   // all waves done reading hs
#pragma unroll
    for (int f = 0; f < 4; ++f) {
      const int j = (w * 4 + f) * 16 + lr;
#pragma unroll
      for (int i = 0; i < 8; ++i) {
        const int bl = bl8 + i;
        __bf16 hb = (__bf16)hnew[f][i];
        hs[bl][j] = hb;
        y[((size_t)(b0 + bl) * TT + tt) * ystride + yoff + j] = hb;
      }
    }
    __syncthreads();   // hs updated for next step
  }

  if (hT_f) {
    for (int i = tid; i < 16 * HH; i += 256) {
      int bl = i >> 9, j = i & (HH - 1);
      float v = (float)hs[bl][j];
      hT_f[(size_t)(b0 + bl) * HH + j] = v;
      hT_b[(size_t)(b0 + bl) * HH + j] = (__bf16)v;
    }
  }
}

// ---------------- decoder GRU cell (one step) ----------------
__global__ __launch_bounds__(256)
void decoder_cell_kernel(const int* __restrict__ target, int l,
                         const __bf16* __restrict__ embed_bf,   // (C, H)
                         const __bf16* __restrict__ wih,        // (3H, H)
                         const __bf16* __restrict__ whh,        // (3H, H)
                         const float* __restrict__ bih, const float* __restrict__ bhh,
                         float* __restrict__ h_f, __bf16* __restrict__ h_b,
                         __bf16* __restrict__ cat_bf)           // (B, 2H), write hn half
{
  __shared__ __bf16 es[16][HH];
  __shared__ __bf16 hs[16][HH];
  const int b0   = blockIdx.x * 16;
  const int tid  = threadIdx.x;
  const int lane = tid & 31;
  const int w    = tid >> 5;
  const int lr   = lane & 15;
  const int kb   = (lane >> 4) * 8;
  const int bl8  = (lane >> 4) * 8;

  for (int i = tid; i < 16 * HH; i += 256) {
    int bl = i >> 9, j = i & (HH - 1);
    int wd = target[(size_t)(b0 + bl) * LL + l];
    es[bl][j] = embed_bf[(size_t)wd * HH + j];
    hs[bl][j] = h_b[(size_t)(b0 + bl) * HH + j];
  }
  __syncthreads();

#pragma unroll
  for (int f = 0; f < 4; ++f) {
    const int c0 = (w * 4 + f) * 16;
    v8f gr = {}, gz = {}, gn = {}, hr = {}, hz = {}, hn = {};
    for (int k0 = 0; k0 < HH; k0 += 32) {
      v16bf ae, ah, bir, biz, bin, bhr, bhz, bhn;
      const __bf16* wir = wih + (size_t)(c0 + lr) * HH + k0 + kb;
      const __bf16* whr = whh + (size_t)(c0 + lr) * HH + k0 + kb;
#pragma unroll
      for (int i = 0; i < 8; ++i) {
        ae[i] = es[lr][k0 + kb + i];      ae[8 + i] = es[lr][k0 + kb + 16 + i];
        ah[i] = hs[lr][k0 + kb + i];      ah[8 + i] = hs[lr][k0 + kb + 16 + i];
        bir[i] = wir[i];                    bir[8 + i] = wir[16 + i];
        biz[i] = wir[(size_t)HH * HH + i];  biz[8 + i] = wir[(size_t)HH * HH + 16 + i];
        bin[i] = wir[(size_t)2*HH*HH + i];  bin[8 + i] = wir[(size_t)2*HH*HH + 16 + i];
        bhr[i] = whr[i];                    bhr[8 + i] = whr[16 + i];
        bhz[i] = whr[(size_t)HH * HH + i];  bhz[8 + i] = whr[(size_t)HH * HH + 16 + i];
        bhn[i] = whr[(size_t)2*HH*HH + i];  bhn[8 + i] = whr[(size_t)2*HH*HH + 16 + i];
      }
      gr = WMMA_BF16(ae, bir, gr);  gz = WMMA_BF16(ae, biz, gz);  gn = WMMA_BF16(ae, bin, gn);
      hr = WMMA_BF16(ah, bhr, hr);  hz = WMMA_BF16(ah, bhz, hz);  hn = WMMA_BF16(ah, bhn, hn);
    }
    const int j = c0 + lr;
#pragma unroll
    for (int i = 0; i < 8; ++i) {
      const int bl = bl8 + i;
      float ir  = gr[i] + bih[j];
      float iz  = gz[i] + bih[HH + j];
      float in_ = gn[i] + bih[2 * HH + j];
      float hrv = hr[i] + bhh[j];
      float hzv = hz[i] + bhh[HH + j];
      float hnv = hn[i] + bhh[2 * HH + j];
      float rg  = 1.f / (1.f + __expf(-(ir + hrv)));
      float zg  = 1.f / (1.f + __expf(-(iz + hzv)));
      float ng  = tanhf(in_ + rg * hnv);
      float hp  = (float)hs[bl][j];
      float hv  = (1.f - zg) * ng + zg * hp;
      h_f[(size_t)(b0 + bl) * HH + j] = hv;
      h_b[(size_t)(b0 + bl) * HH + j] = (__bf16)hv;
      cat_bf[(size_t)(b0 + bl) * (2 * HH) + HH + j] = (__bf16)hv;
    }
  }
}

// ---------------- attention: scores + softmax + context ----------------
__global__ __launch_bounds__(256)
void attention_kernel(const float* __restrict__ enc,  // (B*T, H)
                      const float* __restrict__ h,    // (B, H)
                      __bf16* __restrict__ cat_bf)    // (B, 2H), write ctx half
{
  __shared__ float hn[HH];
  __shared__ float s[TT];
  __shared__ float inv_sum;
  const int b = blockIdx.x, tid = threadIdx.x;
  for (int i = tid; i < HH; i += 256) hn[i] = h[(size_t)b * HH + i];
  __syncthreads();
  if (tid < TT) {
    const float* er = enc + ((size_t)b * TT + tid) * HH;
    float acc = 0.f;
    for (int j = 0; j < HH; ++j) acc += er[j] * hn[j];
    s[tid] = acc;
  }
  __syncthreads();
  if (tid == 0) {
    float m = s[0];
    for (int t = 1; t < TT; ++t) m = fmaxf(m, s[t]);
    float sum = 0.f;
    for (int t = 0; t < TT; ++t) { float e = __expf(s[t] - m); s[t] = e; sum += e; }
    inv_sum = 1.f / sum;
  }
  __syncthreads();
  const float inv = inv_sum;
  for (int j = tid; j < HH; j += 256) {
    float acc = 0.f;
    for (int t = 0; t < TT; ++t) acc += s[t] * enc[((size_t)b * TT + t) * HH + j];
    cat_bf[(size_t)b * (2 * HH) + j] = (__bf16)(acc * inv);
  }
}

// ---------------- log_softmax over C + argmax ----------------
__global__ __launch_bounds__(256)
void logsoftmax_kernel(const float* __restrict__ logits,  // (B, C)
                       float* __restrict__ dout, int l)
{
  __shared__ float rv[256];
  __shared__ int   ri[256];
  const int b = blockIdx.x, tid = threadIdx.x;
  const float* row = logits + (size_t)b * CC;

  float m = -3.4e38f; int mi = 0;
  for (int c = tid; c < CC; c += 256) {
    float v = row[c];
    if (v > m) { m = v; mi = c; }
  }
  rv[tid] = m; ri[tid] = mi;
  __syncthreads();
  for (int st = 128; st > 0; st >>= 1) {
    if (tid < st) {
      if (rv[tid + st] > rv[tid] || (rv[tid + st] == rv[tid] && ri[tid + st] < ri[tid])) {
        rv[tid] = rv[tid + st]; ri[tid] = ri[tid + st];
      }
    }
    __syncthreads();
  }
  const float mx = rv[0]; const int amax = ri[0];
  __syncthreads();

  float sum = 0.f;
  for (int c = tid; c < CC; c += 256) sum += __expf(row[c] - mx);
  rv[tid] = sum;
  __syncthreads();
  for (int st = 128; st > 0; st >>= 1) {
    if (tid < st) rv[tid] += rv[tid + st];
    __syncthreads();
  }
  const float ls = __logf(rv[0]);

  float* pp = dout + ((size_t)b * LL + l) * CC;
  for (int c = tid; c < CC; c += 256) pp[c] = row[c] - mx - ls;
  if (tid == 0) dout[(size_t)BB * LL * CC + (size_t)l * BB + b] = (float)amax;
}

// ---------------- host orchestration ----------------
static inline void cvt(const float* s, __bf16* d, size_t n, hipStream_t st) {
  int g = (int)((n + 1023) / 1024); if (g > 4096) g = 4096;
  cvt_f32_bf16<<<g, 256, 0, st>>>(s, d, n);
}

extern "C" void kernel_launch(void* const* d_in, const int* in_sizes, int n_in,
                              void* d_out, int out_size, void* d_ws, size_t ws_size,
                              hipStream_t stream) {
  const float* data      = (const float*)d_in[0];
  const int*   target    = (const int*)  d_in[1];
  const float* lin_w     = (const float*)d_in[2];
  const float* lin_b     = (const float*)d_in[3];
  const float* wih0      = (const float*)d_in[4];
  const float* whh0      = (const float*)d_in[5];
  const float* bih0      = (const float*)d_in[6];
  const float* bhh0      = (const float*)d_in[7];
  const float* wih1      = (const float*)d_in[8];
  const float* whh1      = (const float*)d_in[9];
  const float* bih1      = (const float*)d_in[10];
  const float* bhh1      = (const float*)d_in[11];
  const float* dwih      = (const float*)d_in[12];
  const float* dwhh      = (const float*)d_in[13];
  const float* dbih      = (const float*)d_in[14];
  const float* dbhh      = (const float*)d_in[15];
  const float* out_w     = (const float*)d_in[16];
  const float* out_b     = (const float*)d_in[17];
  const float* embed     = (const float*)d_in[18];
  float* dout = (float*)d_out;

  // bump allocator over workspace
  char* p = (char*)d_ws;
  auto alloc = [&](size_t bytes) -> void* {
    void* r = (void*)p; p += (bytes + 255) & ~(size_t)255; return r;
  };
  const size_t MT = (size_t)BB * TT;   // 10240 rows
  __bf16* data_bf  = (__bf16*)alloc(MT * INF_ * 2);
  __bf16* linw_bf  = (__bf16*)alloc((size_t)HH * INF_ * 2);
  __bf16* x_bf     = (__bf16*)alloc(MT * HH * 2);
  __bf16* wih0_bf  = (__bf16*)alloc((size_t)2 * H3 * HH * 2);
  __bf16* whh0_bf  = (__bf16*)alloc((size_t)2 * H3 * HH * 2);
  __bf16* wih1_bf  = (__bf16*)alloc((size_t)2 * H3 * 2 * HH * 2);
  __bf16* whh1_bf  = (__bf16*)alloc((size_t)2 * H3 * HH * 2);
  __bf16* dwih_bf  = (__bf16*)alloc((size_t)H3 * HH * 2);
  __bf16* dwhh_bf  = (__bf16*)alloc((size_t)H3 * HH * 2);
  __bf16* outw_bf  = (__bf16*)alloc((size_t)CC * 2 * HH * 2);
  __bf16* emb_bf   = (__bf16*)alloc((size_t)CC * HH * 2);
  float*  gi       = (float*) alloc(MT * H3 * 4);       // reused per layer/direction
  __bf16* y0_bf    = (__bf16*)alloc(MT * 2 * HH * 2);
  __bf16* y1f_bf   = (__bf16*)alloc(MT * HH * 2);
  __bf16* y1b_bf   = (__bf16*)alloc(MT * HH * 2);
  float*  enc_f    = (float*) alloc(MT * HH * 4);
  float*  hdec_f   = (float*) alloc((size_t)BB * HH * 4);
  __bf16* hdec_b   = (__bf16*)alloc((size_t)BB * HH * 2);
  __bf16* cat_bf   = (__bf16*)alloc((size_t)BB * 2 * HH * 2);
  float*  logits   = (float*) alloc((size_t)BB * CC * 4);

  // --- convert fp32 params/activations to bf16 once ---
  cvt(data,  data_bf, MT * INF_, stream);
  cvt(lin_w, linw_bf, (size_t)HH * INF_, stream);
  cvt(wih0,  wih0_bf, (size_t)2 * H3 * HH, stream);
  cvt(whh0,  whh0_bf, (size_t)2 * H3 * HH, stream);
  cvt(wih1,  wih1_bf, (size_t)2 * H3 * 2 * HH, stream);
  cvt(whh1,  whh1_bf, (size_t)2 * H3 * HH, stream);
  cvt(dwih,  dwih_bf, (size_t)H3 * HH, stream);
  cvt(dwhh,  dwhh_bf, (size_t)H3 * HH, stream);
  cvt(out_w, outw_bf, (size_t)CC * 2 * HH, stream);
  cvt(embed, emb_bf,  (size_t)CC * HH, stream);

  // --- encoder linear: x = relu(data @ lin_w^T + b), bf16 out ---
  wmma_gemm_kernel<<<dim3(HH / 64, MT / 128), 256, 0, stream>>>(
      data_bf, linw_bf, lin_b, nullptr, x_bf, (int)MT, HH, INF_, HH, 1);

  // --- GRU layer 0, forward then backward (gi GEMM hoisted out of recurrence) ---
  wmma_gemm_kernel<<<dim3(H3 / 64, MT / 128), 256, 0, stream>>>(
      x_bf, wih0_bf, bih0, gi, nullptr, (int)MT, H3, HH, H3, 0);
  gru_layer_kernel<false><<<BB / 16, 256, 0, stream>>>(
      gi, whh0_bf, bhh0, y0_bf, 2 * HH, 0, nullptr, nullptr);

  wmma_gemm_kernel<<<dim3(H3 / 64, MT / 128), 256, 0, stream>>>(
      x_bf, wih0_bf + (size_t)H3 * HH, bih0 + H3, gi, nullptr, (int)MT, H3, HH, H3, 0);
  gru_layer_kernel<true><<<BB / 16, 256, 0, stream>>>(
      gi, whh0_bf + (size_t)H3 * HH, bhh0 + H3, y0_bf, 2 * HH, HH, nullptr, nullptr);

  // --- GRU layer 1 (input 2H) ---
  wmma_gemm_kernel<<<dim3(H3 / 64, MT / 128), 256, 0, stream>>>(
      y0_bf, wih1_bf, bih1, gi, nullptr, (int)MT, H3, 2 * HH, H3, 0);
  gru_layer_kernel<false><<<BB / 16, 256, 0, stream>>>(
      gi, whh1_bf, bhh1, y1f_bf, HH, 0, nullptr, nullptr);

  wmma_gemm_kernel<<<dim3(H3 / 64, MT / 128), 256, 0, stream>>>(
      y0_bf, wih1_bf + (size_t)H3 * 2 * HH, bih1 + H3, gi, nullptr, (int)MT, H3, 2 * HH, H3, 0);
  gru_layer_kernel<true><<<BB / 16, 256, 0, stream>>>(
      gi, whh1_bf + (size_t)H3 * HH, bhh1 + H3, y1b_bf, HH, 0, hdec_f, hdec_b);

  // encode_out = y1f + y1b
  add_bf16_f32<<<4096, 256, 0, stream>>>(y1f_bf, y1b_bf, enc_f, MT * HH);

  // --- decoder: 10 teacher-forced steps ---
  for (int l = 0; l < LL; ++l) {
    decoder_cell_kernel<<<BB / 16, 256, 0, stream>>>(
        target, l, emb_bf, dwih_bf, dwhh_bf, dbih, dbhh, hdec_f, hdec_b, cat_bf);
    attention_kernel<<<BB, 256, 0, stream>>>(enc_f, hdec_f, cat_bf);
    wmma_gemm_kernel<<<dim3(CC / 64, BB / 128), 256, 0, stream>>>(
        cat_bf, outw_bf, out_b, logits, nullptr, BB, CC, 2 * HH, CC, 0);
    logsoftmax_kernel<<<BB, 256, 0, stream>>>(logits, dout, l);
  }

  (void)in_sizes; (void)n_in; (void)out_size; (void)ws_size;
}
